// MultiheadAttention_64922725646392
// MI455X (gfx1250) — matrix operand order repaired and checked by reading, not compiled, not run on hardware
//
#include <hip/hip_runtime.h>
#include <stdint.h>

#define B_ 4
#define T_ 2048
#define S_ 2048
#define C_ 1024
#define H_ 16
#define D_ 64

typedef __bf16 bf16;
typedef __attribute__((ext_vector_type(16))) bf16 v16bf;
typedef __attribute__((ext_vector_type(8)))  bf16 v8bf;
typedef __attribute__((ext_vector_type(8)))  float v8f;
typedef __attribute__((ext_vector_type(4)))  unsigned int u32x4;
typedef __attribute__((ext_vector_type(8)))  int i32x8;
typedef __attribute__((ext_vector_type(4)))  int i32x4;

__device__ __forceinline__ v8f wmma_bf16(v16bf a, v16bf b, v8f c) {
  // D = A(16x32 bf16) * B(32x16 bf16) + C(16x16 f32)
  return __builtin_amdgcn_wmma_f32_16x16x32_bf16(false, a, false, b, (short)0, c,
                                                 false, false);
}

// 16 contiguous bf16 (32B aligned)
__device__ __forceinline__ v16bf ld16c(const bf16* p) {
  return *reinterpret_cast<const v16bf*>(p);
}

// A-fragment interleave: 8 bf16 at p0 (K=kb..kb+7), 8 at p1 (K=16+kb..16+kb+7)
__device__ __forceinline__ v16bf ld16i(const bf16* p0, const bf16* p1) {
  v8bf a = *reinterpret_cast<const v8bf*>(p0);
  v8bf b = *reinterpret_cast<const v8bf*>(p1);
  v16bf r;
#pragma unroll
  for (int i = 0; i < 8; ++i) { r[i] = a[i]; r[i + 8] = b[i]; }
  return r;
}

// f32 -> bf16 A-fragment (two 8-element halves, K contiguous here)
__device__ __forceinline__ v16bf cvtA(const float* p0, const float* p1) {
  v16bf r;
#pragma unroll
  for (int i = 0; i < 8; ++i) { r[i] = (bf16)p0[i]; r[i + 8] = (bf16)p1[i]; }
  return r;
}

// f32 -> bf16 B-fragment (16 contiguous along K)
__device__ __forceinline__ v16bf cvtB(const float* p) {
  v16bf r;
#pragma unroll
  for (int i = 0; i < 16; ++i) r[i] = (bf16)p[i];
  return r;
}

// ---------------------------------------------------------------------------
// TDM: 2D tile load Global->LDS via Tensor DMA descriptor (ISA ch.7/8).
// Group0: count=1 | lds_addr | global_addr[56:0] | type=2.
// Group1: data_size=2B | tensor_dim0/1 | tile_dim0/1 (tile_dim2=0) | stride0.
// Groups 2/3 zero (<=2D tensor). Tracked with TENSORcnt.
// dims/tiles/stride in elements (bf16). 6-arg builtin (clang-23/therock).
// ---------------------------------------------------------------------------
__device__ __forceinline__ void tdm_load_2d(const bf16* gaddr, unsigned lds_addr,
                                            unsigned dim0, unsigned dim1,
                                            unsigned stride0, unsigned tile0,
                                            unsigned tile1) {
  const uint64_t ga = (uint64_t)(uintptr_t)gaddr;
  u32x4 g0;
  g0[0] = 1u;                                               // count=1 (valid)
  g0[1] = lds_addr;                                         // lds_addr[31:0]
  g0[2] = (unsigned)ga;                                     // global_addr[31:0]
  g0[3] = (unsigned)((ga >> 32) & 0x1FFFFFFu) | (2u << 30); // addr[56:32]|type=2
  i32x8 g1;
  g1[0] = (int)(1u << 16);                        // data_size=1 -> 2 bytes
  g1[1] = (int)((dim0 & 0xFFFFu) << 16);          // tensor_dim0[15:0] @ 63:48
  g1[2] = (int)((dim0 >> 16) | ((dim1 & 0xFFFFu) << 16)); // dim0 hi | dim1 lo
  g1[3] = (int)((dim1 >> 16) | (tile0 << 16));    // dim1 hi | tile_dim0 @127:112
  g1[4] = (int)(tile1 & 0xFFFFu);                 // tile_dim1; tile_dim2=0
  g1[5] = (int)stride0;                           // tensor_dim0_stride[31:0]
  g1[6] = 0;                                      // stride0 hi | stride1 lo
  g1[7] = 0;
  const i32x4 gz4 = {0, 0, 0, 0};
  const i32x8 gz8 = {0, 0, 0, 0, 0, 0, 0, 0};
  __builtin_amdgcn_tensor_load_to_lds(g0, g1, gz4, gz4, gz8, 0);
}

// ---------------------------------------------------------------------------
// Kernel 1: Y[n][o] = (X[n][:] . W[o][:] + bias[o]) * scale   (bf16 output)
// transposedV: write Y as V^T laid out [B][H][D][S] instead of [N][C].
// One wave computes a 32x64 tile. All six fragments of a K-step (2 A, 4 B)
// are loaded/converted into distinct buffers BEFORE the 8-WMMA block, so no
// cvt has a WAR hazard against an in-flight WMMA (kills the v_nop bubbles)
// and the next K-step's loads can issue in the WMMA shadow.
// ---------------------------------------------------------------------------
__global__ __launch_bounds__(128) void qkv_proj_kernel(
    const float* __restrict__ X, const float* __restrict__ W,
    const float* __restrict__ bias, bf16* __restrict__ Y, float scale,
    int transposedV) {
  const int wave  = blockIdx.x * 4 + (threadIdx.x >> 5);
  const int lane  = threadIdx.x & 31;
  const int m     = lane & 15;
  const int hi    = lane >> 4;
  const int kbase = hi * 8;    // unused (A also uses khi)
  const int khi   = hi * 16;   // B-fragment K half
  const int colGroup = wave & 15;  // C/64 = 16 column groups
  const int rowTile  = wave >> 4;  // N/32 = 256 row tiles (32 rows each)
  const int n0 = rowTile * 32;

  const float* xrow0 = X + (size_t)(n0 + m) * C_;
  const float* xrow1 = X + (size_t)(n0 + 16 + m) * C_;
  const float* wrow  = W + (size_t)(colGroup * 64 + m) * C_ + khi;

  v8f zero = {};
  v8f acc[8];  // acc[j] rows n0..n0+15, acc[4+j] rows n0+16..n0+31
#pragma unroll
  for (int j = 0; j < 8; ++j) acc[j] = zero;

  for (int kc = 0; kc < C_; kc += 32) {
    // Stage all fragments for this K-step in distinct register buffers.
    v16bf a0 = cvtA(xrow0 + kc + khi, xrow0 + kc + khi + 8);
    v16bf a1 = cvtA(xrow1 + kc + khi, xrow1 + kc + khi + 8);
    v16bf bw[4];
#pragma unroll
    for (int j = 0; j < 4; ++j) bw[j] = cvtB(wrow + (size_t)(j * 16) * C_ + kc);
    // 8 hazard-free WMMAs (distinct A/B/acc registers).
#pragma unroll
    for (int j = 0; j < 4; ++j) {
      acc[j]     = wmma_bf16(a0, bw[j], acc[j]);
      acc[4 + j] = wmma_bf16(a1, bw[j], acc[4 + j]);
    }
  }

  if (!transposedV) {
#pragma unroll
    for (int half = 0; half < 2; ++half) {
#pragma unroll
      for (int j = 0; j < 4; ++j) {
        const int o    = colGroup * 64 + j * 16 + m;
        const float bj = bias[o];
#pragma unroll
        for (int r = 0; r < 8; ++r) {
          const int n = n0 + 16 * half + r + 8 * hi;
          Y[(size_t)n * C_ + o] = (bf16)((acc[4 * half + j][r] + bj) * scale);
        }
      }
    }
  } else {
    const int bidx  = n0 / S_;
    const int sbase = (n0 % S_) + 8 * hi;
#pragma unroll
    for (int half = 0; half < 2; ++half) {
#pragma unroll
      for (int j = 0; j < 4; ++j) {
        const int o    = colGroup * 64 + j * 16 + m;
        const float bj = bias[o];
        const int hh = o >> 6, dd = o & 63;
        v8bf pack;
#pragma unroll
        for (int r = 0; r < 8; ++r)
          pack[r] = (bf16)((acc[4 * half + j][r] + bj) * scale);
        *reinterpret_cast<v8bf*>(
            Y + ((((size_t)bidx * H_ + hh) * D_ + dd) * S_ + sbase +
                 16 * half)) = pack;
      }
    }
  }
}

// ---------------------------------------------------------------------------
// Kernel 2: flash attention. 8 waves/block share one (b,h); each wave owns
// TWO 16-query tiles (32 queries), so every K/V fragment read from LDS feeds
// two WMMAs. K/V chunks (32 keys) are staged in LDS by the TDM
// (tensor_load_to_lds), double-buffered, synced with s_wait_tensorcnt +
// workgroup barriers. Scores computed transposed (S^T = K*Q^T) so per-query
// softmax stats reduce with a single shfl_xor(16); exp'd fragments repack
// into the B operand of O^T += V^T * P^T.
// ---------------------------------------------------------------------------
__global__ __launch_bounds__(256) void flash_attn_kernel(
    const bf16* __restrict__ Q, const bf16* __restrict__ K,
    const bf16* __restrict__ VT, const unsigned char* __restrict__ mask,
    float* __restrict__ Out) {
  __shared__ bf16 lds_k[2][32 * 64];  // [s within chunk][d]   4KB per buffer
  __shared__ bf16 lds_v[2][64 * 32];  // [d][s within chunk]   4KB per buffer

  const int widx  = threadIdx.x >> 5;
  const int wave  = blockIdx.x * 8 + widx;
  const int lane  = threadIdx.x & 31;
  const int m     = lane & 15;
  const int hi    = lane >> 4;
  const int kbase = hi * 8;
  const int khi   = hi * 16;
  const int qp = wave & 63;   // 64 query-pairs per (b,h)
  const int bh = wave >> 6;   // all 8 waves of a block share bh
  const int b = bh >> 4, h = bh & 15;
  const int t0 = qp * 32;     // this wave: queries t0 .. t0+31

  // Q B-fragments for both 16-query tiles, d interleaved (table order).
  v16bf qf[2][2];
#pragma unroll
  for (int qi = 0; qi < 2; ++qi) {
    const bf16* qrow = Q + (size_t)(b * T_ + t0 + 16 * qi + m) * C_ + h * D_;
    qf[qi][0] = ld16i(qrow + kbase, qrow + 16 + kbase);
    qf[qi][1] = ld16i(qrow + 32 + kbase, qrow + 48 + kbase);
  }

  const bf16* kb_ = K + (size_t)b * S_ * C_ + h * D_;    // [s][d], stride C
  const bf16* vb_ = VT + (size_t)(b * H_ + h) * D_ * S_; // [d][s], stride S
  const unsigned char* mrow = mask + (size_t)b * S_;

  // Prologue: wave 0 launches the TDM for chunk 0 into buffer 0.
  if (widx == 0) {
    tdm_load_2d(kb_, (unsigned)(uintptr_t)&lds_k[0][0], 64, 32, C_, 64, 32);
    tdm_load_2d(vb_, (unsigned)(uintptr_t)&lds_v[0][0], 32, 64, S_, 32, 64);
  }

  float mstat[2] = {-3.0e38f, -3.0e38f};
  float lsum[2]  = {0.f, 0.f};
  v8f zero = {};
  v8f O[2][4];
#pragma unroll
  for (int qi = 0; qi < 2; ++qi)
#pragma unroll
    for (int j = 0; j < 4; ++j) O[qi][j] = zero;

  for (int s0 = 0; s0 < S_; s0 += 32) {
    const int p = (s0 >> 5) & 1;
    __syncthreads();  // everyone done reading buffer 1-p from the previous iter
    if (widx == 0) {
      if (s0 + 32 < S_) {
        const int sn = s0 + 32;
        tdm_load_2d(kb_ + (size_t)sn * C_,
                    (unsigned)(uintptr_t)&lds_k[1 - p][0], 64, 32, C_, 64, 32);
        tdm_load_2d(vb_ + sn, (unsigned)(uintptr_t)&lds_v[1 - p][0], 32, 64,
                    S_, 32, 64);
        __builtin_amdgcn_s_wait_tensorcnt(2);  // chunk p's two loads complete
      } else {
        __builtin_amdgcn_s_wait_tensorcnt(0);
      }
    }
    __syncthreads();  // buffer p now valid for all waves

    // K A-fragments from LDS, shared by both query tiles.
    v16bf ka[2][2];
    {
      const bf16* kr0 = &lds_k[p][(size_t)m * 64];
      ka[0][0] = ld16i(kr0 + kbase, kr0 + 16 + kbase);
      ka[0][1] = ld16i(kr0 + 32 + kbase, kr0 + 48 + kbase);
      const bf16* kr1 = &lds_k[p][(size_t)(16 + m) * 64];
      ka[1][0] = ld16i(kr1 + kbase, kr1 + 16 + kbase);
      ka[1][1] = ld16i(kr1 + 32 + kbase, kr1 + 48 + kbase);
    }
    // key mask words (shared by both query tiles)
    const uint64_t mw0 = *reinterpret_cast<const uint64_t*>(mrow + s0 + 8 * hi);
    const uint64_t mw1 =
        *reinterpret_cast<const uint64_t*>(mrow + s0 + 16 + 8 * hi);

    v16bf pf[2];
    float corr[2];
#pragma unroll
    for (int qi = 0; qi < 2; ++qi) {
      // Two 16x16 transposed score tiles: rows = keys, cols = queries.
      v8f sc0 = zero, sc1 = zero;
      sc0 = wmma_bf16(ka[0][0], qf[qi][0], sc0);
      sc0 = wmma_bf16(ka[0][1], qf[qi][1], sc0);
      sc1 = wmma_bf16(ka[1][0], qf[qi][0], sc1);
      sc1 = wmma_bf16(ka[1][1], qf[qi][1], sc1);
      if (mw0 | mw1) {
#pragma unroll
        for (int r = 0; r < 8; ++r) {
          if ((mw0 >> (8 * r)) & 1) sc0[r] = -3.0e38f;
          if ((mw1 >> (8 * r)) & 1) sc1[r] = -3.0e38f;
        }
      }
      // online softmax stats for this lane's query column (t = m)
      float mx = fmaxf(sc0[0], sc1[0]);
#pragma unroll
      for (int r = 1; r < 8; ++r) mx = fmaxf(mx, fmaxf(sc0[r], sc1[r]));
      mx = fmaxf(mx, __shfl_xor(mx, 16, 32));
      const float mnew = fmaxf(mstat[qi], mx);
      corr[qi] = __expf(mstat[qi] - mnew);

      float e0[8], e1[8], ls = 0.f;
#pragma unroll
      for (int r = 0; r < 8; ++r) {
        e0[r] = __expf(sc0[r] - mnew);
        e1[r] = __expf(sc1[r] - mnew);
        ls += e0[r] + e1[r];
      }
      ls += __shfl_xor(ls, 16, 32);
      lsum[qi] = lsum[qi] * corr[qi] + ls;
      mstat[qi] = mnew;

      // Repack P^T into a B-fragment: lane needs K = 16*hi + idx; the
      // missing 8 rows live in lane^16.
#pragma unroll
      for (int r = 0; r < 8; ++r) {
        const float x0 = __shfl_xor(e0[r], 16, 32);
        const float x1 = __shfl_xor(e1[r], 16, 32);
        pf[qi][r]     = (bf16)(hi ? x1 : e0[r]);
        pf[qi][r + 8] = (bf16)(hi ? e1[r] : x0);
      }
    }

    // O^T rescale + accumulate; each V fragment feeds both query tiles.
#pragma unroll
    for (int j = 0; j < 4; ++j) {
      const bf16* vr = &lds_v[p][(size_t)(j * 16 + m) * 32];
      const v16bf va = ld16i(vr + khi, vr + khi + 8);
#pragma unroll
      for (int qi = 0; qi < 2; ++qi) {
#pragma unroll
        for (int r = 0; r < 8; ++r) O[qi][j][r] *= corr[qi];
        O[qi][j] = wmma_bf16(va, pf[qi], O[qi][j]);
      }
    }
  }

#pragma unroll
  for (int qi = 0; qi < 2; ++qi) {
    const float inv = 1.0f / lsum[qi];
    float* orow =
        Out + (size_t)(b * T_ + t0 + 16 * qi + m) * C_ + h * D_ + 8 * hi;
#pragma unroll
    for (int j = 0; j < 4; ++j) {
      float4 lo = make_float4(O[qi][j][0] * inv, O[qi][j][1] * inv,
                              O[qi][j][2] * inv, O[qi][j][3] * inv);
      float4 hi4 = make_float4(O[qi][j][4] * inv, O[qi][j][5] * inv,
                               O[qi][j][6] * inv, O[qi][j][7] * inv);
      float* p4 = orow + j * 16;
      *reinterpret_cast<float4*>(p4) = lo;
      *reinterpret_cast<float4*>(p4 + 4) = hi4;
    }
  }
}

// ---------------------------------------------------------------------------
extern "C" void kernel_launch(void* const* d_in, const int* in_sizes, int n_in,
                              void* d_out, int out_size, void* d_ws,
                              size_t ws_size, hipStream_t stream) {
  (void)in_sizes; (void)n_in; (void)out_size; (void)ws_size;
  const float* query = (const float*)d_in[0];
  const float* key   = (const float*)d_in[1];
  const float* value = (const float*)d_in[2];
  const unsigned char* kmask = (const unsigned char*)d_in[3];
  const float* Wq = (const float*)d_in[4];
  const float* bq = (const float*)d_in[5];
  const float* Wk = (const float*)d_in[6];
  const float* bk = (const float*)d_in[7];
  const float* Wv = (const float*)d_in[8];
  const float* bv = (const float*)d_in[9];

  // Workspace: bf16 q [B,T,C], k [B,S,C], v^T [B,H,D,S]  (48 MB total)
  bf16* qb = (bf16*)d_ws;
  bf16* kb = qb + (size_t)B_ * T_ * C_;
  bf16* vT = kb + (size_t)B_ * S_ * C_;

  const int projBlocks = (B_ * T_ / 32) * (C_ / 64) / 4;  // 1024 blocks x 4 waves
  qkv_proj_kernel<<<projBlocks, 128, 0, stream>>>(query, Wq, bq, qb,
                                                  0.125f /* 1/sqrt(D) */, 0);
  qkv_proj_kernel<<<projBlocks, 128, 0, stream>>>(key, Wk, bk, kb, 1.0f, 0);
  qkv_proj_kernel<<<projBlocks, 128, 0, stream>>>(value, Wv, bv, vT, 1.0f, 1);

  const int attnBlocks = (B_ * H_ * (T_ / 32)) / 8;  // 512 blocks x 8 waves
  flash_attn_kernel<<<attnBlocks, 256, 0, stream>>>(qb, kb, vT, kmask,
                                                    (float*)d_out);
}
